// HypernetFullyConnectedFlexiLayerSizeArchCore_71124658422107
// MI455X (gfx1250) — compile-verified, matrix operand order checked by reading, not run on hardware
//
#include <hip/hip_runtime.h>
#include <math.h>

#define IN_FEATURES 512
#define NUM_CASES   64
#define N_POINTS    131072
#define HIDDEN      512
#define XDIM        513   // 512 feats + cid (stored as float)

typedef __attribute__((ext_vector_type(16))) _Float16 v16h;
typedef __attribute__((ext_vector_type(8)))  float    v8f;
typedef __attribute__((ext_vector_type(4)))  float    v4f;   // native vector for nontemporal b128

// ---------------------------------------------------------------------------
// K0: zero the segmented-sum accumulators (S: 64x512, cnt: 64)
// ---------------------------------------------------------------------------
__global__ void zero_kernel(float* __restrict__ buf, int n) {
    int i = blockIdx.x * blockDim.x + threadIdx.x;
    if (i < n) buf[i] = 0.0f;
}

// ---------------------------------------------------------------------------
// K1: segmented sum. grid = (256 point-chunks, 4 feature slices), block = 128.
// LDS partials: 64 cases x 128 feats (32 KB). Accumulation uses no-return
// ds_add_f32 (atomicAdd on shared): one DS op per element, no wait chain.
// Feature loads are non-temporal (269 MB streamed once, don't pollute L2).
// ---------------------------------------------------------------------------
__global__ void segsum_kernel(const float* __restrict__ x,
                              float* __restrict__ S,      // 64*512
                              float* __restrict__ cnt) {  // 64
    __shared__ float lsum[NUM_CASES * 128];
    __shared__ float lcnt[NUM_CASES];
    const int tid = threadIdx.x;
    for (int i = tid; i < NUM_CASES * 128; i += 128) lsum[i] = 0.0f;
    if (tid < NUM_CASES) lcnt[tid] = 0.0f;
    __syncthreads();

    const int slice = blockIdx.y;            // 0..3
    const int f0    = slice * 128;
    const int chunk = N_POINTS / gridDim.x;  // 512
    const int p0    = blockIdx.x * chunk;

#pragma unroll 4
    for (int p = p0; p < p0 + chunk; ++p) {
        const float* row = x + (size_t)p * XDIM;
        const int c = (int)row[IN_FEATURES];                        // cid
        const float v = __builtin_nontemporal_load(row + f0 + tid); // coalesced 512B
        atomicAdd(&lsum[c * 128 + tid], v);                         // ds_add_f32
        if (slice == 0 && tid == 0) lcnt[c] += 1.0f;
    }
    __syncthreads();

    for (int c = 0; c < NUM_CASES; ++c)
        atomicAdd(&S[c * HIDDEN + f0 + tid], lsum[c * 128 + tid]);
    if (slice == 0 && tid < NUM_CASES)
        atomicAdd(&cnt[tid], lcnt[tid]);
}

// ---------------------------------------------------------------------------
// K2: h0 = S / max(cnt, 1)
// ---------------------------------------------------------------------------
__global__ void mean_kernel(const float* __restrict__ S,
                            const float* __restrict__ cnt,
                            float* __restrict__ h0) {
    int i = blockIdx.x * blockDim.x + threadIdx.x;
    if (i < NUM_CASES * HIDDEN) {
        float c = cnt[i >> 9];                 // i / 512
        c = (c < 1.0f) ? 1.0f : c;
        h0[i] = S[i] / c;
    }
}

// ---------------------------------------------------------------------------
// K3: per-row weight-norm scale: scale[j] = g[j] * rsqrt(sum_k V[j,k]^2)
// One 128-thread block per row.
// ---------------------------------------------------------------------------
__global__ void rownorm_kernel(const float* __restrict__ V,
                               const float* __restrict__ g,
                               float* __restrict__ scale) {
    __shared__ float red[128];
    const int row = blockIdx.x;
    float s = 0.0f;
    for (int k = threadIdx.x; k < HIDDEN; k += 128) {
        float v = V[(size_t)row * HIDDEN + k];
        s += v * v;
    }
    red[threadIdx.x] = s;
    __syncthreads();
    for (int off = 64; off > 0; off >>= 1) {
        if (threadIdx.x < off) red[threadIdx.x] += red[threadIdx.x + off];
        __syncthreads();
    }
    if (threadIdx.x == 0) scale[row] = g[row] * rsqrtf(red[0]);
}

// ---------------------------------------------------------------------------
// Load 16 f16 K-values for one lane per the documented 16-bit A/B layout:
//   elem e (0..15):  K = (e & 7) + (e >= 8 ? 16 : 0) + (lane >= 16 ? 8 : 0)
// The two 8-element halves are contiguous in memory -> two b128 loads each.
// ---------------------------------------------------------------------------
__device__ inline v16h load_row16_f16(const float* __restrict__ p, int lane) {
    const float* q = p + ((lane >= 16) ? 8 : 0);
    v16h r;
#pragma unroll
    for (int e = 0; e < 8; ++e) r[e]     = (_Float16)q[e];
#pragma unroll
    for (int e = 0; e < 8; ++e) r[8 + e] = (_Float16)q[16 + e];
    return r;
}

// ---------------------------------------------------------------------------
// K4: hout(64x512) = act( scale[j] * (hin(64x512) @ V(512x512)^T)[i,j] + b[j] )
// One wave (32 lanes) per 16x16 output tile, 128 tiles total.
// grid = 32 blocks x 128 threads (4 waves/block). EXEC all ones -> WMMA legal.
// SILU / HAS_SCALE are compile-time: no runtime branches in the epilogue.
// ---------------------------------------------------------------------------
template <bool SILU, bool HAS_SCALE>
__global__ void wn_gemm_kernel(const float* __restrict__ hin,
                               const float* __restrict__ V,
                               const float* __restrict__ scale,
                               const float* __restrict__ bias,
                               float* __restrict__ hout) {
    const int wave = blockIdx.x * (blockDim.x >> 5) + (threadIdx.x >> 5); // 0..127
    const int lane = threadIdx.x & 31;
    const int mt = wave & 3;        // 4 tiles along M (64/16)
    const int nt = wave >> 2;       // 32 tiles along N (512/16)
    const int m0 = mt * 16;
    const int j0 = nt * 16;
    const int l15 = lane & 15;

    v8f acc = {};
    const float* arow = hin + (size_t)(m0 + l15) * HIDDEN;  // A: M = lane&15
    const float* brow = V   + (size_t)(j0 + l15) * HIDDEN;  // B: N = lane&15 (out = h @ V^T)

    for (int kk = 0; kk < HIDDEN; kk += 32) {
        v16h a = load_row16_f16(arow + kk, lane);
        v16h b = load_row16_f16(brow + kk, lane);
        acc = __builtin_amdgcn_wmma_f32_16x16x32_f16(
            /*neg_a=*/false, a, /*neg_b=*/false, b,
            /*c_mod=*/(short)0, acc, /*reuse_a=*/false, /*reuse_b=*/false);
    }

    const int n  = j0 + l15;
    const float sc = HAS_SCALE ? scale[n] : 1.0f;
    const float bs = bias[n];
    const int mhi = (lane >= 16) ? 8 : 0;
#pragma unroll
    for (int r = 0; r < 8; ++r) {
        const int m = m0 + r + mhi;            // C/D layout: VGPR r -> M = r (+8)
        float v = acc[r] * sc + bs;
        if (SILU) {
            // silu(v) = v * sigmoid(v); fast reciprocal (v_rcp_f32)
            v = v * __builtin_amdgcn_rcpf(1.0f + __expf(-v));
        }
        hout[(size_t)m * HIDDEN + n] = v;
    }
}

// ---------------------------------------------------------------------------
// K5: out[n,:] = h3[cid[n],:]. One block per point. h3 rows and out rows are
// 16B-aligned -> v4f (b128) non-temporal stores on the 256 MB write stream.
// ---------------------------------------------------------------------------
__global__ void scatter_kernel(const float* __restrict__ x,
                               const float* __restrict__ h3,
                               float* __restrict__ out) {
    const int p = blockIdx.x;
    const int c = (int)x[(size_t)p * XDIM + IN_FEATURES];
    const v4f* __restrict__ src = (const v4f*)(h3 + (size_t)c * HIDDEN);
    v4f* __restrict__ dst = (v4f*)(out + (size_t)p * HIDDEN);
    const int t = threadIdx.x;                 // 128 threads x v4f = 512 floats
    __builtin_nontemporal_store(src[t], dst + t);
}

// ---------------------------------------------------------------------------
// Orchestration
// ---------------------------------------------------------------------------
extern "C" void kernel_launch(void* const* d_in, const int* in_sizes, int n_in,
                              void* d_out, int out_size, void* d_ws, size_t ws_size,
                              hipStream_t stream) {
    (void)in_sizes; (void)n_in; (void)out_size; (void)ws_size;

    const float* x  = (const float*)d_in[0];
    const float* V0 = (const float*)d_in[1];
    const float* g0 = (const float*)d_in[2];
    const float* b0 = (const float*)d_in[3];
    const float* V1 = (const float*)d_in[4];
    const float* g1 = (const float*)d_in[5];
    const float* b1 = (const float*)d_in[6];
    const float* Wf = (const float*)d_in[7];
    const float* bf = (const float*)d_in[8];
    float* out = (float*)d_out;

    // Workspace layout (floats), ~660 KB total
    float* ws = (float*)d_ws;
    float* S      = ws;                 // 64*512
    float* cnt    = S   + 64 * 512;     // 64
    float* h0     = cnt + 64;           // 64*512
    float* h1     = h0  + 64 * 512;
    float* h2     = h1  + 64 * 512;
    float* h3     = h2  + 64 * 512;
    float* scale0 = h3  + 64 * 512;     // 512
    float* scale1 = scale0 + 512;       // 512

    // K0: zero accumulators (S + cnt)
    zero_kernel<<<(64 * 512 + 64 + 255) / 256, 256, 0, stream>>>(S, 64 * 512 + 64);

    // K3: weight-norm scales (independent of K1/K2)
    rownorm_kernel<<<512, 128, 0, stream>>>(V0, g0, scale0);
    rownorm_kernel<<<512, 128, 0, stream>>>(V1, g1, scale1);

    // K1: segmented sums. 256 point-chunks x 4 feature slices.
    {
        dim3 grid(256, 4);
        segsum_kernel<<<grid, 128, 0, stream>>>(x, S, cnt);
    }

    // K2: means
    mean_kernel<<<(64 * 512 + 255) / 256, 256, 0, stream>>>(S, cnt, h0);

    // K4 x3: MLP layers on 64x512 via WMMA
    wn_gemm_kernel<true,  true ><<<32, 128, 0, stream>>>(h0, V0, scale0, b0, h1);
    wn_gemm_kernel<true,  true ><<<32, 128, 0, stream>>>(h1, V1, scale1, b1, h2);
    wn_gemm_kernel<false, false><<<32, 128, 0, stream>>>(h2, Wf, nullptr,  bf, h3);

    // K5: scatter rows back to all points
    scatter_kernel<<<N_POINTS, 128, 0, stream>>>(x, h3, out);
}
